// BasicConceptQuantizationV3_77318001262769
// MI455X (gfx1250) — compile-verified, matrix operand order checked by reading, not run on hardware
//
#include <hip/hip_runtime.h>
#include <hip/hip_bf16.h>
#include <stdint.h>
#include <stddef.h>

// ---------------------------------------------------------------------------
// Types for CDNA5 WMMA (wave32): A/B = 16 bf16 per lane (8 VGPRs), C/D = 8 f32
// ---------------------------------------------------------------------------
typedef __attribute__((ext_vector_type(16))) __bf16 v16bf;
typedef __attribute__((ext_vector_type(8)))  float  v8f;
typedef __attribute__((ext_vector_type(4)))  unsigned int v4u;
typedef __attribute__((ext_vector_type(8)))  int  v8i;
typedef __attribute__((ext_vector_type(4)))  int  v4i;

union Frag16 {
    uint4 q[2];   // two 16-byte chunks == 8 VGPRs
    v16bf v;
};

__device__ __forceinline__ unsigned short f32_to_bf16_rne(float f) {
    unsigned int u = __float_as_uint(f);
    u += 0x7FFFu + ((u >> 16) & 1u);     // round-to-nearest-even
    return (unsigned short)(u >> 16);
}

#define BM 128
#define BN 128
#define BKK 32
#define LDSS 40   // padded LDS row stride (bf16): 80B -> conflict-free b128 reads

#if defined(__has_builtin)
#if __has_builtin(__builtin_amdgcn_tensor_load_to_lds) && \
    __has_builtin(__builtin_amdgcn_s_wait_tensorcnt)
#define USE_TDM 1
#endif
#endif

#ifdef USE_TDM
// ---------------------------------------------------------------------------
// TDM: async-DMA one 128-row x 32-elem bf16 tile (row stride ld_elems) into
// LDS at lds_off, with hardware padding 64B data -> +16B pad per row, which
// reproduces the LDSS=40 layout. D# per CDNA5 ISA ch.8 (§8.3/8.4).
// Tensor dims set to huge so OOB never fires (tiles are fully in-bounds).
// This toolchain exposes the 6-arg builtin:
//   (uint32x4 g0, int32x8 g1, int32x4 g2, int32x4 g3, int32x8 extra, i32 cpol)
// Groups 2/3 are unused for 2-D tensors -> zero-filled.
// ---------------------------------------------------------------------------
__device__ __forceinline__ void tdm_load_tile_2d(const unsigned short* gptr,
                                                 unsigned lds_off, int ld_elems)
{
    unsigned long long ga = (unsigned long long)(uintptr_t)gptr;
    v4u g0;
    g0[0] = 1u;                                   // count=1 (valid), user mode
    g0[1] = lds_off;                              // lds_addr (bytes)
    g0[2] = (unsigned)(ga & 0xFFFFFFFFu);         // global_addr[31:0]
    g0[3] = (unsigned)((ga >> 32) & 0x1FFFFFFu)   // global_addr[56:32]
          | (2u << 30);                           // type = 2 ("image")
    v8i g1;
    g1[0] = (1 << 16)                             // data_size = 2 bytes (bf16)
          | (1 << 20)                             // pad_enable
          | (3 << 22)                             // pad_interval: 16 DWORDs (64B)
          | (3 << 25);                            // pad_amount: 4 DWORDs (16B)
    g1[1] = 0;                                    // atomic barrier addr 0, dim0 lo16=0
    g1[2] = 0x4000;                               // tensor_dim0 = 1<<30, dim1 lo16=0
    g1[3] = 0x4000 | (BKK << 16);                 // tensor_dim1 = 1<<30, tile_dim0=32
    g1[4] = BM;                                   // tile_dim1 = 128 rows, tile_dim2=0
    g1[5] = ld_elems;                             // tensor_dim0_stride lo32 (elems)
    g1[6] = 0;                                    // stride hi16 / dim1_stride lo16
    g1[7] = 0;
    v4i z4 = {0, 0, 0, 0};
    v8i z8 = {0, 0, 0, 0, 0, 0, 0, 0};
    __builtin_amdgcn_tensor_load_to_lds(g0, g1, z4, z4, z8, 0);
}
#endif

// ---------------------------------------------------------------------------
// WMMA GEMM, C[M,N] = alpha * A[M,K] * B^T  (B stored [N,K] row-major, bf16)
// Block tile 128x128x32, 256 threads (8 waves), each wave computes 64x32.
// Preconditions (arranged by caller): M % 128 == 0, K % 32 == 0, and the B
// buffer has at least ceil(N/128)*128 rows (zero-padded) -> no load guards.
// Stores are guarded by the true N. Optional f32/bf16 outputs, optional bias.
// ---------------------------------------------------------------------------
__global__ __launch_bounds__(256) void wmma_gemm_bt(
    const unsigned short* __restrict__ A,   // [M,K] bf16
    const unsigned short* __restrict__ B,   // [>=N(pad),K] bf16 (BT layout)
    float* __restrict__ Cf,                 // optional [M,N] f32 out
    unsigned short* __restrict__ Cbf,       // optional [M,N] bf16 out
    const float* __restrict__ bias,         // optional [N]
    int M, int N, int K, float alpha)
{
#ifdef USE_TDM
    __shared__ unsigned short sA[2][BM * LDSS];
    __shared__ unsigned short sB[2][BN * LDSS];
#else
    __shared__ unsigned short sA[1][BM * LDSS];
    __shared__ unsigned short sB[1][BN * LDSS];
#endif

    const int tid  = threadIdx.x;
    const int lane = tid & 31;
    const int wave = tid >> 5;
    const int l15  = lane & 15;
    const int lhi  = lane >> 4;

    const int blockM = blockIdx.y * BM;
    const int blockN = blockIdx.x * BN;
    const int waveM  = (wave >> 2) * 64;   // 0 or 64
    const int waveN  = (wave & 3) * 32;    // 0,32,64,96

    v8f acc[4][2];
#pragma unroll
    for (int t = 0; t < 4; ++t)
#pragma unroll
        for (int u = 0; u < 2; ++u)
#pragma unroll
            for (int j = 0; j < 8; ++j) acc[t][u][j] = 0.0f;

    const int nIter = K / BKK;

#ifdef USE_TDM
    const unsigned ldsA0 = (unsigned)(uintptr_t)&sA[0][0];
    const unsigned ldsA1 = (unsigned)(uintptr_t)&sA[1][0];
    const unsigned ldsB0 = (unsigned)(uintptr_t)&sB[0][0];
    const unsigned ldsB1 = (unsigned)(uintptr_t)&sB[1][0];
    if (tid < 32) {   // wave 0 drives the Tensor Data Mover
        tdm_load_tile_2d(A + (size_t)blockM * K, ldsA0, K);
        tdm_load_tile_2d(B + (size_t)blockN * K, ldsB0, K);
    }
#endif

    for (int it = 0; it < nIter; ++it) {
        const int k0 = it * BKK;
#ifdef USE_TDM
        const int buf = it & 1;
        if (tid < 32) __builtin_amdgcn_s_wait_tensorcnt(0);  // buf `it` landed
        __syncthreads();                                     // publish to WG
        if (tid < 32 && (it + 1) < nIter) {                  // prefetch buf it+1
            const int k1 = k0 + BKK;
            tdm_load_tile_2d(A + (size_t)blockM * K + k1, buf ? ldsA0 : ldsA1, K);
            tdm_load_tile_2d(B + (size_t)blockN * K + k1, buf ? ldsB0 : ldsB1, K);
        }
#else
        const int buf = 0;
        // ---- manual staging: unguarded straight-line b128 loads ----
#pragma unroll
        for (int i = 0; i < 2; ++i) {
            int idx   = tid + i * 256;      // 0..511
            int row   = idx >> 2;           // 0..127
            int chunk = (idx & 3) * 8;      // bf16 offset within row
            uint4 za = *(const uint4*)(A + (size_t)(blockM + row) * K + k0 + chunk);
            *(uint4*)(&sA[0][row * LDSS + chunk]) = za;
            uint4 zb = *(const uint4*)(B + (size_t)(blockN + row) * K + k0 + chunk);
            *(uint4*)(&sB[0][row * LDSS + chunk]) = zb;
        }
        __syncthreads();
#endif

        // ---- fragments: ISA 16-bit 16x32 striping, two b128 per lane ----
        Frag16 a[4], b[2];
#pragma unroll
        for (int t = 0; t < 4; ++t) {
            const unsigned short* p = &sA[buf][(waveM + t * 16 + l15) * LDSS + lhi * 8];
            a[t].q[0] = *(const uint4*)p;          // K = lhi*8 .. +7
            a[t].q[1] = *(const uint4*)(p + 16);   // K = lhi*8+16 .. +23
        }
#pragma unroll
        for (int u = 0; u < 2; ++u) {
            const unsigned short* p = &sB[buf][(waveN + u * 16 + l15) * LDSS + lhi * 8];
            b[u].q[0] = *(const uint4*)p;
            b[u].q[1] = *(const uint4*)(p + 16);
        }

        // ---- 8 WMMAs per K-step: 4 A-frags x 2 B-frags reuse ----
#pragma unroll
        for (int t = 0; t < 4; ++t)
#pragma unroll
            for (int u = 0; u < 2; ++u)
                acc[t][u] = __builtin_amdgcn_wmma_f32_16x16x32_bf16(
                    false, a[t].v, false, b[u].v, (short)0, acc[t][u],
                    false, false);

#ifndef USE_TDM
        __syncthreads();
#endif
    }

    // ---- epilogue: C/D layout -> m = j + 8*(lane>=16), n = lane&15 ----
#pragma unroll
    for (int t = 0; t < 4; ++t) {
#pragma unroll
        for (int u = 0; u < 2; ++u) {
#pragma unroll
            for (int j = 0; j < 8; ++j) {
                int m = blockM + waveM + t * 16 + lhi * 8 + j;
                int n = blockN + waveN + u * 16 + l15;
                if (m < M && n < N) {
                    float v = acc[t][u][j] * alpha;
                    if (bias) v += bias[n];
                    if (Cf)  Cf[(size_t)m * N + n] = v;
                    if (Cbf) Cbf[(size_t)m * N + n] = f32_to_bf16_rne(v);
                }
            }
        }
    }
}

// ---------------------------------------------------------------------------
// Row L2-normalize: one 256-thread block per row of length D.
// ---------------------------------------------------------------------------
__global__ __launch_bounds__(256) void normalize_rows(
    const float* __restrict__ in, float* __restrict__ out_f32,
    unsigned short* __restrict__ out_bf16, int D)
{
    __shared__ float red[256];
    const int row = blockIdx.x;
    const float* r = in + (size_t)row * D;
    float s = 0.f;
    for (int i = threadIdx.x; i < D; i += 256) { float v = r[i]; s += v * v; }
    red[threadIdx.x] = s;
    __syncthreads();
    for (int o = 128; o > 0; o >>= 1) {
        if (threadIdx.x < o) red[threadIdx.x] += red[threadIdx.x + o];
        __syncthreads();
    }
    const float rn = rsqrtf(red[0]);
    for (int i = threadIdx.x; i < D; i += 256) {
        float v = r[i] * rn;
        if (out_f32)  out_f32[(size_t)row * D + i]  = v;
        if (out_bf16) out_bf16[(size_t)row * D + i] = f32_to_bf16_rne(v);
    }
}

// ---------------------------------------------------------------------------
// Elementwise f32 -> bf16, and bf16 zero-fill (for row padding)
// ---------------------------------------------------------------------------
__global__ void convert_f32_bf16(const float* __restrict__ in,
                                 unsigned short* __restrict__ out, size_t n)
{
    size_t i = (size_t)blockIdx.x * blockDim.x + threadIdx.x;
    if (i < n) out[i] = f32_to_bf16_rne(in[i]);
}

__global__ void fill_zero_bf16(unsigned short* __restrict__ p, size_t n)
{
    size_t i = (size_t)blockIdx.x * blockDim.x + threadIdx.x;
    if (i < n) p[i] = 0;
}

// ---------------------------------------------------------------------------
// Transposing convert: in f32 [R,C] -> out bf16 [C,R]  (32x32 LDS tiles)
// ---------------------------------------------------------------------------
__global__ __launch_bounds__(256) void transpose_f32_bf16(
    const float* __restrict__ in, unsigned short* __restrict__ out, int R, int C)
{
    __shared__ unsigned short tile[32][33];
    const int bx = blockIdx.x * 32;   // col base in input
    const int by = blockIdx.y * 32;   // row base in input
    const int tx = threadIdx.x & 31;
    const int ty = threadIdx.x >> 5;  // 32x8
#pragma unroll
    for (int i = 0; i < 32; i += 8) {
        int r = by + ty + i, c = bx + tx;
        unsigned short v = 0;
        if (r < R && c < C) v = f32_to_bf16_rne(in[(size_t)r * C + c]);
        tile[ty + i][tx] = v;
    }
    __syncthreads();
#pragma unroll
    for (int i = 0; i < 32; i += 8) {
        int r = bx + ty + i, c = by + tx;     // out is [C,R]
        if (r < C && c < R) out[(size_t)r * R + c] = tile[tx][ty + i];
    }
}

// ---------------------------------------------------------------------------
// Sparsemax over rows of length 512: one 512-thread block per row.
// Bitonic sort (descending) + inclusive scan + support count, per reference.
// ---------------------------------------------------------------------------
__global__ __launch_bounds__(512) void sparsemax_512(
    const float* __restrict__ z, float* __restrict__ out_f32,
    unsigned short* __restrict__ out_bf16)
{
    const int C = 512;
    __shared__ float s[512];
    __shared__ float cs[512];
    __shared__ int kz;
    const int row = blockIdx.x;
    const int tid = threadIdx.x;

    const float zv = z[(size_t)row * C + tid];
    s[tid] = zv;
    __syncthreads();

    // bitonic sort, descending
    for (int k = 2; k <= 512; k <<= 1) {
        for (int j = k >> 1; j > 0; j >>= 1) {
            int ixj = tid ^ j;
            if (ixj > tid) {
                float a = s[tid], b = s[ixj];
                bool desc = ((tid & k) == 0);
                if (desc ? (a < b) : (a > b)) { s[tid] = b; s[ixj] = a; }
            }
            __syncthreads();
        }
    }

    // inclusive scan (Hillis-Steele)
    cs[tid] = s[tid];
    __syncthreads();
    for (int off = 1; off < 512; off <<= 1) {
        float add = (tid >= off) ? cs[tid - off] : 0.f;
        __syncthreads();
        cs[tid] += add;
        __syncthreads();
    }

    if (tid == 0) kz = 0;
    __syncthreads();
    int sup = (1.0f + (float)(tid + 1) * s[tid] > cs[tid]) ? 1 : 0;
    atomicAdd(&kz, sup);
    __syncthreads();

    const int k_z = kz;
    const float tau = (cs[k_z - 1] - 1.0f) / (float)k_z;
    const float o = fmaxf(zv - tau, 0.0f);
    out_f32[(size_t)row * C + tid]  = o;
    out_bf16[(size_t)row * C + tid] = f32_to_bf16_rne(o);
}

// ---------------------------------------------------------------------------
// Launch pipeline.
// ---------------------------------------------------------------------------
extern "C" void kernel_launch(void* const* d_in, const int* in_sizes, int n_in,
                              void* d_out, int out_size, void* d_ws, size_t ws_size,
                              hipStream_t stream)
{
    (void)in_sizes; (void)n_in; (void)out_size; (void)ws_size;
    const int Bb = 16384, D = 1024, Cc = 512, Ncls = 1000;
    const int NclsPad = 1024;   // fc_w rows padded so GEMM loads need no guards

    const float* x        = (const float*)d_in[0];   // [B,D]
    const float* concepts = (const float*)d_in[1];   // [C,D]
    const float* qt       = (const float*)d_in[2];   // [D,D]
    const float* kt       = (const float*)d_in[3];   // [D,D]
    const float* fcw      = (const float*)d_in[4];   // [N,D]
    const float* fcb      = (const float*)d_in[5];   // [N]

    float* out        = (float*)d_out;
    float* out_logits = out;                              // [B,Ncls]
    float* out_attn   = out + (size_t)Bb * Ncls;          // [B,Cc]
    float* out_csim   = out_attn + (size_t)Bb * Cc;       // [Cc,Cc]

    char* w = (char*)d_ws;
    auto alloc = [&](size_t bytes) -> char* {
        char* p = w;
        w += (bytes + 255) & ~(size_t)255;
        return p;
    };
    float*          c_f32   = (float*)         alloc((size_t)Cc * D * 4);
    unsigned short* c_bf    = (unsigned short*)alloc((size_t)Cc * D * 2);
    unsigned short* cT_bf   = (unsigned short*)alloc((size_t)D * Cc * 2);   // [D,C]
    unsigned short* qtT_bf  = (unsigned short*)alloc((size_t)D * D * 2);    // [D,D]^T
    unsigned short* ktT_bf  = (unsigned short*)alloc((size_t)D * D * 2);
    unsigned short* x_bf    = (unsigned short*)alloc((size_t)Bb * D * 2);
    unsigned short* q_bf    = (unsigned short*)alloc((size_t)Bb * D * 2);
    unsigned short* k_bf    = (unsigned short*)alloc((size_t)Cc * D * 2);   // [C,D]
    float*          z_f32   = (float*)         alloc((size_t)Bb * Cc * 4);  // logits
    unsigned short* attn_bf = (unsigned short*)alloc((size_t)Bb * Cc * 2);
    float*          sum_f32 = (float*)         alloc((size_t)Bb * D * 4);
    unsigned short* sumn_bf = (unsigned short*)alloc((size_t)Bb * D * 2);
    unsigned short* fcw_bf  = (unsigned short*)alloc((size_t)NclsPad * D * 2);

    const float inv_sqrt_d = 0.03125f;   // 1/sqrt(1024)

    // 1) c = normalize(concepts)  -> f32 + bf16
    normalize_rows<<<Cc, 256, 0, stream>>>(concepts, c_f32, c_bf, D);

    // 2) pre-transpose weights to BT layout (bf16)
    transpose_f32_bf16<<<dim3(D / 32, D / 32), 256, 0, stream>>>(qt, qtT_bf, D, D);
    transpose_f32_bf16<<<dim3(D / 32, D / 32), 256, 0, stream>>>(kt, ktT_bf, D, D);
    transpose_f32_bf16<<<dim3(D / 32, Cc / 32), 256, 0, stream>>>(c_f32, cT_bf, Cc, D);

    // 3) stage activations / fc weights as bf16; zero-pad fc_w rows 1000..1023
    {
        size_t n = (size_t)Bb * D;
        convert_f32_bf16<<<(unsigned)((n + 255) / 256), 256, 0, stream>>>(x, x_bf, n);
        size_t nf = (size_t)Ncls * D;
        convert_f32_bf16<<<(unsigned)((nf + 255) / 256), 256, 0, stream>>>(fcw, fcw_bf, nf);
        size_t np = (size_t)(NclsPad - Ncls) * D;
        fill_zero_bf16<<<(unsigned)((np + 255) / 256), 256, 0, stream>>>(fcw_bf + nf, np);
    }

    // 4) q = x @ Wq  -> bf16  [B,D]
    wmma_gemm_bt<<<dim3(D / BN, Bb / BM), 256, 0, stream>>>(
        x_bf, qtT_bf, nullptr, q_bf, nullptr, Bb, D, D, 1.0f);

    // 5) k = c @ Wk  -> bf16  [C,D]
    wmma_gemm_bt<<<dim3(D / BN, Cc / BM), 256, 0, stream>>>(
        c_bf, ktT_bf, nullptr, k_bf, nullptr, Cc, D, D, 1.0f);

    // 6) z = (q @ k^T) / sqrt(D)  -> f32 [B,C]   (k row-major IS the BT layout)
    wmma_gemm_bt<<<dim3(Cc / BN, Bb / BM), 256, 0, stream>>>(
        q_bf, k_bf, z_f32, nullptr, nullptr, Bb, Cc, D, inv_sqrt_d);

    // 7) attn = sparsemax(z)  -> f32 output + bf16 copy
    sparsemax_512<<<Bb, 512, 0, stream>>>(z_f32, out_attn, attn_bf);

    // 8) summary = attn @ c  -> f32 [B,D]   (B = c^T in BT layout)
    wmma_gemm_bt<<<dim3(D / BN, Bb / BM), 256, 0, stream>>>(
        attn_bf, cT_bf, sum_f32, nullptr, nullptr, Bb, D, Cc, 1.0f);

    // 9) summary = normalize(summary)  -> bf16
    normalize_rows<<<Bb, 256, 0, stream>>>(sum_f32, nullptr, sumn_bf, D);

    // 10) outputs = summary @ fc_w^T + fc_b  -> f32 [B,Ncls]
    wmma_gemm_bt<<<dim3((Ncls + BN - 1) / BN, Bb / BM), 256, 0, stream>>>(
        sumn_bf, fcw_bf, out_logits, nullptr, fcb, Bb, Ncls, D, 1.0f);

    // 11) concept_similarity = c @ c^T  -> f32 [C,C]
    wmma_gemm_bt<<<dim3(Cc / BN, Cc / BM), 256, 0, stream>>>(
        c_bf, c_bf, out_csim, nullptr, nullptr, Cc, Cc, D, 1.0f);
}
